// LovaszSoftmax_72447508349621
// MI455X (gfx1250) — compile-verified, matrix operand order checked by reading, not run on hardware
//
#include <hip/hip_runtime.h>
#include <stdint.h>

typedef unsigned int uint;

typedef uint u32x4 __attribute__((ext_vector_type(4)));
typedef int  i32x8 __attribute__((ext_vector_type(8)));
typedef int  i32x4 __attribute__((ext_vector_type(4)));
typedef int  v8i   __attribute__((ext_vector_type(8)));

#define BATCH    8
#define NPIX     262144
#define NCLS_TOT 21
#define NCLS     20
#define NSEG     160        /* BATCH * NCLS */
#define TILE     4096
#define NBLK     64         /* NPIX / TILE */
#define BINS     1024       /* 10-bit digits, 3 passes cover the <2^30 keys */
#define CHUNK    8192
#define NCHUNK   32         /* NPIX / CHUNK */

#if defined(__AMDGCN__) && __has_builtin(__builtin_amdgcn_tensor_load_to_lds) && __has_builtin(__builtin_amdgcn_s_wait_tensorcnt)
#define USE_TDM 1
#else
#define USE_TDM 0
#endif

// ---------------------------------------------------------------------------
// Zero the per-batch valid-pixel counters (accumulated atomically in build).
// ---------------------------------------------------------------------------
__global__ void lz_init(uint* validCount) {
    if (threadIdx.x < BATCH) validCount[threadIdx.x] = 0u;
}

// ---------------------------------------------------------------------------
// Fused softmax + key build.  Key for valid pixel: ((bits(e) & ~1) | fg) + 2,
// which is monotone in e (errors are in [0,1] so bits are order-preserving),
// carries fg in the LSB, and is < 2^30.  Invalid pixels get key 0 (sorts to
// the tail of the descending traversal with e=0, fg=0).
// ---------------------------------------------------------------------------
__global__ __launch_bounds__(256) void lz_build_keys(
    const float* __restrict__ logits, const int* __restrict__ labels,
    uint* __restrict__ keys, uint* __restrict__ validCount)
{
    int p = blockIdx.x * 256 + threadIdx.x;          // grid exactly covers B*N
    int b = p / NPIX, n = p % NPIX;
    const float* lg = logits + (size_t)p * NCLS_TOT;

    float v[NCLS_TOT];
    float m = -1e30f;
    #pragma unroll
    for (int c = 0; c < NCLS_TOT; ++c) { v[c] = lg[c]; m = fmaxf(m, v[c]); }
    float s = 0.f;
    #pragma unroll
    for (int c = 0; c < NCLS_TOT; ++c) { v[c] = __expf(v[c] - m); s += v[c]; }
    float inv = 1.0f / s;

    int  lab   = labels[p];
    bool valid = (lab != 0);
    uint bal = (uint)__ballot(valid);                // wave32 ballot
    if ((threadIdx.x & 31) == 0) atomicAdd(&validCount[b], (uint)__popc(bal));

    #pragma unroll
    for (int c = 1; c < NCLS_TOT; ++c) {
        float pr   = v[c] * inv;
        uint  isfg = (lab == c) ? 1u : 0u;
        float e    = fabsf((float)isfg - pr);
        uint  kb   = 0u;
        if (valid) kb = ((__float_as_uint(e) & ~1u) | isfg) + 2u;
        keys[(size_t)(b * NCLS + (c - 1)) * NPIX + n] = kb;
    }
}

// ---------------------------------------------------------------------------
// Radix pass 1/3: per-block digit histograms.  Tile is staged into LDS by
// the Tensor Data Mover (one D# issued by wave 0), waited with TENSORcnt.
// ---------------------------------------------------------------------------
__global__ __launch_bounds__(256) void lz_radix_hist(
    const uint* __restrict__ in, uint* __restrict__ gHist, int shift)
{
    __shared__ uint tile[TILE];
    __shared__ uint hcnt[BINS];
    int blk = blockIdx.x, seg = blockIdx.y, t = threadIdx.x;
    const uint* src = in + (size_t)seg * NPIX + (size_t)blk * TILE;

    for (int d = t; d < BINS; d += 256) hcnt[d] = 0u;

#if USE_TDM
    if (t < 32) {                                    // wave 0 issues the TDM op
        u32x4 g0; i32x8 g1;
        i32x4 gz4 = {0, 0, 0, 0};
        i32x8 gz8 = {0, 0, 0, 0, 0, 0, 0, 0};
        uint64_t ga  = (uint64_t)(uintptr_t)src;
        uint     lds = (uint)(uintptr_t)&tile[0];    // low 32 bits = LDS offset
        g0.x = 1u;                                   // count=1 valid descriptor
        g0.y = lds;                                  // lds_addr
        g0.z = (uint)(ga & 0xffffffffu);             // global_addr[31:0]
        g0.w = (uint)((ga >> 32) & 0x01ffffffu) | (2u << 30); // addr hi | type=2
        g1[0] = (int)0x00020000u;                    // data_size=2 (4B), no mcast
        g1[1] = (int)((TILE & 0xffffu) << 16);       // tensor_dim0 lo16
        g1[2] = (int)(((TILE >> 16) & 0xffffu) | (1u << 16)); // dim0 hi | dim1=1
        g1[3] = (int)((TILE & 0xffffu) << 16);       // dim1 hi(0) | tile_dim0
        g1[4] = 1;                                   // tile_dim1=1, tile_dim2=0
        g1[5] = (int)TILE;                           // tensor_dim0_stride lo
        g1[6] = 0; g1[7] = 0;
        __builtin_amdgcn_tensor_load_to_lds(g0, g1, gz4, gz4, gz8, 0);
        __builtin_amdgcn_s_wait_tensorcnt(0);
    }
    __syncthreads();
#else
    for (int j = t; j < TILE; j += 256) tile[j] = src[j];
    __syncthreads();
#endif

    #pragma unroll
    for (int j = 0; j < TILE / 256; ++j) {
        uint k = tile[j * 256 + t];
        atomicAdd(&hcnt[(k >> shift) & (BINS - 1)], 1u);
    }
    __syncthreads();
    for (int d = t; d < BINS; d += 256)
        gHist[((size_t)seg * BINS + d) * NBLK + blk] = hcnt[d];
}

// ---------------------------------------------------------------------------
// Radix pass 2/3: in-place exclusive scan over blocks per (seg,digit), and
// per-digit totals.
// ---------------------------------------------------------------------------
__global__ __launch_bounds__(256) void lz_radix_scan_blocks(
    uint* __restrict__ gHist, uint* __restrict__ gTot)
{
    int idx = blockIdx.x * 256 + threadIdx.x;
    if (idx >= NSEG * BINS) return;
    uint* p = gHist + (size_t)idx * NBLK;
    uint s = 0;
    #pragma unroll 4
    for (int k = 0; k < NBLK; ++k) { uint v = p[k]; p[k] = s; s += v; }
    gTot[idx] = s;
}

// Exclusive scan across 1024 digit totals, one workgroup per segment.
__global__ __launch_bounds__(1024) void lz_radix_scan_digits(
    const uint* __restrict__ gTot, uint* __restrict__ gBase)
{
    __shared__ uint sb[2][BINS];
    int seg = blockIdx.x, d = threadIdx.x;
    sb[0][d] = gTot[(size_t)seg * BINS + d];
    int src = 0;
    for (int off = 1; off < BINS; off <<= 1) {
        __syncthreads();
        uint v = sb[src][d];
        if (d >= off) v += sb[src][d - off];
        sb[src ^ 1][d] = v; src ^= 1;
    }
    __syncthreads();
    gBase[(size_t)seg * BINS + d] = (d == 0) ? 0u : sb[src][d - 1];
}

// ---------------------------------------------------------------------------
// Radix pass 3/3: stable scatter.  Stability within the block comes from
// (iteration, wave, lane) ordering: 10 wave32 ballots build the exact
// same-digit mask, rank = popcount below my lane; cross-wave offsets via a
// per-iteration LDS wave-count table; running per-digit bases via LDS atomics.
// ---------------------------------------------------------------------------
__global__ __launch_bounds__(256) void lz_radix_scatter(
    const uint* __restrict__ in, uint* __restrict__ out,
    const uint* __restrict__ gHist, const uint* __restrict__ gBase, int shift)
{
    __shared__ uint baseLds[BINS];
    __shared__ uint waveCnt[8][BINS];
    int blk = blockIdx.x, seg = blockIdx.y, t = threadIdx.x;
    int w = t >> 5, lane = t & 31;
    uint lmask_lt = (1u << lane) - 1u;
    const uint* src = in + (size_t)seg * NPIX + (size_t)blk * TILE;

    for (int d = t; d < BINS; d += 256)
        baseLds[d] = gBase[(size_t)seg * BINS + d] +
                     gHist[((size_t)seg * BINS + d) * NBLK + blk];
    __syncthreads();

    for (int it = 0; it < TILE / 256; ++it) {
        uint k = src[it * 256 + t];
        if (it + 1 < TILE / 256)
            __builtin_prefetch(&src[(it + 1) * 256 + t], 0, 0);
        uint d = (k >> shift) & (BINS - 1);

        #pragma unroll
        for (int z = 0; z < (8 * BINS) / 256; ++z)
            ((uint*)waveCnt)[z * 256 + t] = 0u;
        __syncthreads();

        uint mask = 0xffffffffu;
        #pragma unroll
        for (int bpos = 0; bpos < 10; ++bpos) {
            uint bal = (uint)__ballot((d >> bpos) & 1);
            mask &= ((d >> bpos) & 1) ? bal : ~bal;
        }
        uint rank = (uint)__popc(mask & lmask_lt);
        uint cnt  = (uint)__popc(mask);
        if (rank == 0) waveCnt[w][d] = cnt;
        __syncthreads();

        uint woff = 0;
        for (int w2 = 0; w2 < w; ++w2) woff += waveCnt[w2][d];
        uint pos = baseLds[d] + woff + rank;
        out[(size_t)seg * NPIX + pos] = k;
        __syncthreads();
        if (rank == 0) atomicAdd(&baseLds[d], cnt);
        __syncthreads();
    }
}

// ---------------------------------------------------------------------------
// Per-chunk fg counts via the matrix unit: fg bytes (0/1) x all-ones B with
// V_WMMA_I32_16X16X64_IU8.  With B==1 everywhere, the D matrix holds row sums
// replicated across 16 columns, so sum(all D slots) == 16 * popcount.
// ---------------------------------------------------------------------------
__global__ __launch_bounds__(256) void lz_fg_partials(
    const uint* __restrict__ keys, uint* __restrict__ gPart)
{
    __shared__ uint wsum[8];
    int q = blockIdx.x, seg = blockIdx.y, t = threadIdx.x;
    int w = t >> 5, lane = t & 31;
    const uint* src = keys + (size_t)seg * NPIX + (size_t)q * CHUNK + (size_t)w * 1024;

    v8i a = {0, 0, 0, 0, 0, 0, 0, 0};
    #pragma unroll
    for (int j = 0; j < 32; ++j) {
        uint k = src[j * 32 + lane];                 // coalesced across lanes
        a[j >> 2] |= (int)((k & 1u) << ((j & 3) * 8));
    }
    v8i bOnes = {0x01010101, 0x01010101, 0x01010101, 0x01010101,
                 0x01010101, 0x01010101, 0x01010101, 0x01010101};
    v8i c0 = {0, 0, 0, 0, 0, 0, 0, 0};
    v8i dm = __builtin_amdgcn_wmma_i32_16x16x64_iu8(false, a, false, bOnes,
                                                    c0, false, false);
    int s = dm[0] + dm[1] + dm[2] + dm[3] + dm[4] + dm[5] + dm[6] + dm[7];
    #pragma unroll
    for (int off = 16; off > 0; off >>= 1) s += __shfl_xor(s, off, 32);
    if (lane == 0) wsum[w] = ((uint)s) >> 4;         // /16 -> this wave's count
    __syncthreads();
    if (t == 0) {
        uint tot = 0;
        #pragma unroll
        for (int i = 0; i < 8; ++i) tot += wsum[i];
        gPart[(size_t)seg * NCHUNK + q] = tot;
    }
}

// Exclusive fg prefix per chunk in DESCENDING traversal order (= suffix over
// ascending chunk index), plus total fg G per segment.  Tiny: 160 x 32.
__global__ void lz_scan_partials(const uint* __restrict__ gPart,
                                 uint* __restrict__ gExcl, float* __restrict__ gG)
{
    int s = blockIdx.x * blockDim.x + threadIdx.x;
    if (s >= NSEG) return;
    uint run = 0;
    for (int q = NCHUNK - 1; q >= 0; --q) {
        gExcl[(size_t)s * NCHUNK + q] = run;
        run += gPart[(size_t)s * NCHUNK + q];
    }
    gG[s] = (float)run;
}

// ---------------------------------------------------------------------------
// Per-element Lovasz terms.  Descending rank i = NPIX-1-a (a = ascending
// index).  Each element needs only its exclusive fg prefix Sx:
//   term = e * ( J(i, Sx+fg) - J(i-1, Sx) ),  J(i,S) = 1 - (G-S)/(G+i+1-S).
// Reverse-exclusive fg scan across the 256 threads in LDS, serial within the
// 32 contiguous elements per thread.  Deterministic block reduce to a slot.
// ---------------------------------------------------------------------------
__global__ __launch_bounds__(256) void lz_terms(
    const uint* __restrict__ keys, const uint* __restrict__ gExcl,
    const float* __restrict__ gG, float* __restrict__ gBlockTerms)
{
    __shared__ uint  cb[2][256];
    __shared__ float red[256];
    int q = blockIdx.x, seg = blockIdx.y, t = threadIdx.x;
    const uint* src = keys + (size_t)seg * NPIX + (size_t)q * CHUNK;

    uint k[32];
    const uint4* p4 = (const uint4*)(src + t * 32);
    #pragma unroll
    for (int j = 0; j < 8; ++j) {
        uint4 v = p4[j];
        k[j * 4 + 0] = v.x; k[j * 4 + 1] = v.y;
        k[j * 4 + 2] = v.z; k[j * 4 + 3] = v.w;
    }
    uint ct = 0;
    #pragma unroll
    for (int j = 0; j < 32; ++j) ct += k[j] & 1u;

    cb[0][255 - t] = ct;                 // reversed so forward scan = suffix
    int sb = 0;
    for (int off = 1; off < 256; off <<= 1) {
        __syncthreads();
        uint v = cb[sb][t];
        if (t >= off) v += cb[sb][t - off];
        cb[sb ^ 1][t] = v; sb ^= 1;
    }
    __syncthreads();
    uint sfx = cb[sb][255 - t] - ct;     // fg count in threads after me

    uint  Sx0 = gExcl[(size_t)seg * NCHUNK + q] + sfx;
    float G   = gG[seg];
    size_t aBase = (size_t)q * CHUNK + (size_t)t * 32;

    float term = 0.f; uint run = 0;
    #pragma unroll
    for (int j = 31; j >= 0; --j) {      // descending order within my run
        uint kk = k[j];
        uint fg = kk & 1u;
        uint Sx = Sx0 + run;
        long i  = (long)(NPIX - 1) - (long)(aBase + j);
        float e  = (kk >= 2u) ? __uint_as_float((kk - 2u) & ~1u) : 0.f;
        float Sf = (float)(Sx + fg);
        float fi = (float)i;
        float Ji = 1.f - (G - Sf) / (G + fi + 1.f - Sf);
        float Jp = (i == 0) ? 0.f
                 : (1.f - (G - (float)Sx) / (G + fi - (float)Sx));
        term += e * (Ji - Jp);
        run  += fg;
    }
    red[t] = term;
    for (int s2 = 128; s2 > 0; s2 >>= 1) {
        __syncthreads();
        if (t < s2) red[t] += red[t + s2];
    }
    if (t == 0) gBlockTerms[(size_t)seg * NCHUNK + q] = red[0];
}

// Final deterministic reduction with the per-batch include mask.
__global__ void lz_final(const float* __restrict__ gBlockTerms,
                         const uint* __restrict__ validCount, float* __restrict__ out)
{
    __shared__ float ts[NSEG];
    int t = threadIdx.x;
    if (t < NSEG) {
        float s = 0.f;
        for (int q = 0; q < NCHUNK; ++q) s += gBlockTerms[(size_t)t * NCHUNK + q];
        ts[t] = s;
    }
    __syncthreads();
    if (t == 0) {
        int nb = 0; float loss = 0.f;
        for (int b = 0; b < BATCH; ++b) if (validCount[b] >= 2u) nb++;
        for (int s2 = 0; s2 < NSEG; ++s2)
            if (validCount[s2 / NCLS] >= 2u) loss += ts[s2];
        int cnt = nb * NCLS;
        out[0] = loss / (float)(cnt > 0 ? cnt : 1);
    }
}

// ---------------------------------------------------------------------------
extern "C" void kernel_launch(void* const* d_in, const int* in_sizes, int n_in,
                              void* d_out, int out_size, void* d_ws, size_t ws_size,
                              hipStream_t stream)
{
    (void)in_sizes; (void)n_in; (void)out_size; (void)ws_size;
    const float* logits = (const float*)d_in[0];
    const int*   labels = (const int*)d_in[1];
    float*       out    = (float*)d_out;

    char* w = (char*)d_ws;
    auto carve = [&](size_t bytes) -> void* {
        void* p = (void*)w;
        w += (bytes + 255) & ~(size_t)255;
        return p;
    };
    uint*  keysA = (uint*) carve((size_t)NSEG * NPIX * 4);
    uint*  keysB = (uint*) carve((size_t)NSEG * NPIX * 4);
    uint*  gHist = (uint*) carve((size_t)NSEG * BINS * NBLK * 4);
    uint*  gTot  = (uint*) carve((size_t)NSEG * BINS * 4);
    uint*  gBase = (uint*) carve((size_t)NSEG * BINS * 4);
    uint*  gPart = (uint*) carve((size_t)NSEG * NCHUNK * 4);
    uint*  gExcl = (uint*) carve((size_t)NSEG * NCHUNK * 4);
    float* gG    = (float*)carve((size_t)NSEG * 4);
    float* gBT   = (float*)carve((size_t)NSEG * NCHUNK * 4);
    uint*  vcnt  = (uint*) carve(256);

    lz_init<<<1, 64, 0, stream>>>(vcnt);
    lz_build_keys<<<(BATCH * NPIX) / 256, 256, 0, stream>>>(logits, labels, keysA, vcnt);

    uint* inK = keysA; uint* outK = keysB;
    const int shifts[3] = {0, 10, 20};
    for (int p = 0; p < 3; ++p) {
        lz_radix_hist<<<dim3(NBLK, NSEG), 256, 0, stream>>>(inK, gHist, shifts[p]);
        lz_radix_scan_blocks<<<(NSEG * BINS + 255) / 256, 256, 0, stream>>>(gHist, gTot);
        lz_radix_scan_digits<<<NSEG, BINS, 0, stream>>>(gTot, gBase);
        lz_radix_scatter<<<dim3(NBLK, NSEG), 256, 0, stream>>>(inK, outK, gHist, gBase, shifts[p]);
        uint* tmp = inK; inK = outK; outK = tmp;
    }
    // After A->B, B->A, A->B the sorted keys are in `inK` (== keysB).
    lz_fg_partials<<<dim3(NCHUNK, NSEG), 256, 0, stream>>>(inK, gPart);
    lz_scan_partials<<<1, 256, 0, stream>>>(gPart, gExcl, gG);
    lz_terms<<<dim3(NCHUNK, NSEG), 256, 0, stream>>>(inK, gExcl, gG, gBT);
    lz_final<<<1, 256, 0, stream>>>(gBT, vcnt, out);
}
